// SelfAttn_10359461118467
// MI455X (gfx1250) — matrix operand order repaired and checked by reading, not compiled, not run on hardware
//
#include <hip/hip_runtime.h>
#include <hip/hip_bf16.h>

// ---------------------------------------------------------------------------
// Problem constants (match reference)
// ---------------------------------------------------------------------------
#define BB   16
#define TT   1024
#define DD   512
#define NG   4        // kv groups
#define NPG  2        // query heads per group
#define HH   64       // head dim
#define GHH  256      // NG*HH  (k/v width)
#define NW   1024     // qkv concat width (512 + 256 + 256)

typedef __attribute__((ext_vector_type(16))) _Float16 v16h;
typedef __attribute__((ext_vector_type(8)))  _Float16 v8h;
typedef __attribute__((ext_vector_type(8)))  float    v8f;

// ---------------------------------------------------------------------------
// gfx1250 async global->LDS path (ASYNCcnt), with portable fallback.
// Builtin signature (from compiler diagnostic): params are pointers to
// 'int __attribute__((vector_size(16)))' in AS1 (global) / AS3 (LDS).
// ---------------------------------------------------------------------------
#if defined(__has_builtin)
#if __has_builtin(__builtin_amdgcn_global_load_async_to_lds_b128)
#define HAVE_ASYNC_LDS 1
#endif
#endif

#ifdef HAVE_ASYNC_LDS
typedef int v4i_ __attribute__((vector_size(4 * sizeof(int))));
typedef __attribute__((address_space(1))) v4i_ as1_v4i;
typedef __attribute__((address_space(3))) v4i_ as3_v4i;
#define ASYNC_CP16(l, g)                                                     \
  __builtin_amdgcn_global_load_async_to_lds_b128(                            \
      (as1_v4i*)(g), (as3_v4i*)(l), 0, 0)
#if __has_builtin(__builtin_amdgcn_s_wait_asynccnt)
#define WAIT_ASYNC(n) __builtin_amdgcn_s_wait_asynccnt(n)
#else
#define WAIT_ASYNC(n) asm volatile("s_wait_asynccnt %0" ::"i"(n) : "memory")
#endif
#else
#define ASYNC_CP16(l, g) (*(v8h*)(l) = *(const v8h*)(g))
#define WAIT_ASYNC(n) do {} while (0)
#endif

// ---------------------------------------------------------------------------
// WMMA operand loaders implementing the CDNA5 VGPR layouts (CDNA5 ISA §7.12.2)
// A 16x32 f16: lane m=lane&15; kb=(lane>>4)*8; elems 0..7 -> K=kb+0..7,
//              elems 8..15 -> K=kb+16..23.
// B 32x16 f16 from N-major source Bt[n][k]: lane n=lane&15; kb=(lane>>4)*8;
//              elems 0..7 -> K=kb+0..7, elems 8..15 -> K=kb+16..23.
// C/D 16x16 f32: elem r -> row r + 8*(lane>>4), col = lane&15.
// ---------------------------------------------------------------------------
__device__ __forceinline__ v16h cat16(v8h lo, v8h hi) {
  v16h r;
#pragma unroll
  for (int i = 0; i < 8; ++i) { r[i] = lo[i]; r[8 + i] = hi[i]; }
  return r;
}

__device__ __forceinline__ v16h load_a16(const _Float16* src, int ld) {
  int lane = threadIdx.x & 31;
  int m  = lane & 15;
  int kb = (lane >> 4) << 3;
  const _Float16* r = src + (size_t)m * ld + kb;
  return cat16(*(const v8h*)(r), *(const v8h*)(r + 16));
}

__device__ __forceinline__ v16h load_b16(const _Float16* src, int ld) {
  int lane = threadIdx.x & 31;
  int n  = lane & 15;
  int kb = (lane >> 4) << 3;
  const _Float16* r = src + (size_t)n * ld + kb;
  return cat16(*(const v8h*)(r), *(const v8h*)(r + 16));
}

__device__ __forceinline__ v8f wmma16(v16h a, v16h b, v8f c) {
  return __builtin_amdgcn_wmma_f32_16x16x32_f16(false, a, false, b,
                                                (short)0, c, false, false);
}

// ---------------------------------------------------------------------------
// 1) elementwise casts / tables
// ---------------------------------------------------------------------------
__global__ void cast_x_kernel(const float* __restrict__ x,
                              _Float16* __restrict__ xh) {
  size_t i = (size_t)blockIdx.x * blockDim.x + threadIdx.x;
  if (i < (size_t)BB * TT * DD) xh[i] = (_Float16)x[i];
}

// Wt[n][k] : n<512 -> Wq[k][n]; n<768 -> Wk[k][n-512]; else Wv[k][n-768]
__global__ void cast_w_kernel(const float* __restrict__ Wq,
                              const float* __restrict__ Wk,
                              const float* __restrict__ Wv,
                              _Float16* __restrict__ wt) {
  int idx = blockIdx.x * blockDim.x + threadIdx.x;
  if (idx >= NW * DD) return;
  int n = idx >> 9, k = idx & 511;
  float v;
  if (n < 512)      v = Wq[k * 512 + n];
  else if (n < 768) v = Wk[k * 256 + (n - 512)];
  else              v = Wv[k * 256 + (n - 768)];
  wt[idx] = (_Float16)v;
}

// rope tables in f64, matching the reference's numpy path
__global__ void tables_kernel(float* __restrict__ cq, float* __restrict__ sq,
                              float* __restrict__ ck, float* __restrict__ sk) {
  int idx = blockIdx.x * blockDim.x + threadIdx.x;
  if (idx >= TT * (256 + 128)) return;
  int t, i, dd, off; float *cd, *sd;
  if (idx < TT * 256) { t = idx >> 8; i = idx & 255; dd = DD;  cd = cq; sd = sq; off = t * 256 + i; }
  else { int r = idx - TT * 256; t = r >> 7; i = r & 127; dd = GHH; cd = ck; sd = sk; off = t * 128 + i; }
  double theta = pow(10000.0, -2.0 * ((double)i - 1.0)) / (double)dd;
  double ang = (double)t * theta, s, c;
  sincos(ang, &s, &c);
  cd[off] = (float)c; sd[off] = (float)s;
}

// ---------------------------------------------------------------------------
// 2) fused QKV projection: [B*T,512] x [512,1024] -> f16 [B*T,1024]
//    one wave per 16(M) x 64(N) block: A-tile reused across 4 N-tiles,
//    4 independent WMMA accumulator chains (64 WMMAs / wave).
// ---------------------------------------------------------------------------
__global__ __launch_bounds__(128) void proj_kernel(const _Float16* __restrict__ xh,
                                                   const _Float16* __restrict__ wt,
                                                   _Float16* __restrict__ qkv) {
  int wave = blockIdx.x * (blockDim.x >> 5) + (threadIdx.x >> 5);
  int ntb = wave & 15;       // 16 blocks of 64 over N=1024
  int mt  = wave >> 4;       // 1024 tiles over M=16384
  const _Float16* arow = xh + (size_t)mt * 16 * DD;
  const _Float16* brow = wt + (size_t)ntb * 64 * DD;
  v8f acc0 = {}, acc1 = {}, acc2 = {}, acc3 = {};
#pragma unroll
  for (int k0 = 0; k0 < DD; k0 += 32) {
    v16h a = load_a16(arow + k0, DD);
    acc0 = wmma16(a, load_b16(brow + (size_t)0 * 16 * DD + k0, DD), acc0);
    acc1 = wmma16(a, load_b16(brow + (size_t)1 * 16 * DD + k0, DD), acc1);
    acc2 = wmma16(a, load_b16(brow + (size_t)2 * 16 * DD + k0, DD), acc2);
    acc3 = wmma16(a, load_b16(brow + (size_t)3 * 16 * DD + k0, DD), acc3);
  }
  int lane = threadIdx.x & 31, half = lane >> 4, col = lane & 15;
  _Float16* ob = qkv + (size_t)mt * 16 * NW + ntb * 64;
#pragma unroll
  for (int e = 0; e < 8; ++e) {
    _Float16* orow = ob + (size_t)(e + 8 * half) * NW + col;
    orow[0]  = (_Float16)acc0[e];
    orow[16] = (_Float16)acc1[e];
    orow[32] = (_Float16)acc2[e];
    orow[48] = (_Float16)acc3[e];
  }
}

// ---------------------------------------------------------------------------
// 3) rope + repack into WMMA-friendly layouts
// ---------------------------------------------------------------------------
__global__ void rope_q_kernel(const _Float16* __restrict__ qkv,
                              const float* __restrict__ cq,
                              const float* __restrict__ sq,
                              _Float16* __restrict__ qh) {
  size_t idx = (size_t)blockIdx.x * blockDim.x + threadIdx.x;
  if (idx >= (size_t)BB * TT * 256) return;
  int i = (int)(idx & 255);
  size_t bt = idx >> 8;
  int t = (int)(bt & (TT - 1)); int b = (int)(bt >> 10);
  const _Float16* row = qkv + bt * NW;
  float x0 = (float)row[2 * i], x1 = (float)row[2 * i + 1];
  float c = cq[t * 256 + i], s = sq[t * 256 + i];
  float oe =  x0 * c + x1 * s;
  float oo = -x0 * s + x1 * c;
  int c0 = 2 * i;
  int g = c0 >> 7, p = (c0 >> 6) & 1, d = c0 & 63;
  _Float16* dst = qh + ((((size_t)b * NG + g) * NPG + p) * TT + t) * HH + d;
  dst[0] = (_Float16)oe;
  dst[1] = (_Float16)oo;
}

__global__ void rope_k_kernel(const _Float16* __restrict__ qkv,
                              const float* __restrict__ ck,
                              const float* __restrict__ sk,
                              _Float16* __restrict__ kh) {
  size_t idx = (size_t)blockIdx.x * blockDim.x + threadIdx.x;
  if (idx >= (size_t)BB * TT * 128) return;
  int i = (int)(idx & 127);
  size_t bt = idx >> 7;
  int t = (int)(bt & (TT - 1)); int b = (int)(bt >> 10);
  const _Float16* row = qkv + bt * NW + 512;
  float x0 = (float)row[2 * i], x1 = (float)row[2 * i + 1];
  float c = ck[t * 128 + i], s = sk[t * 128 + i];
  float oe =  x0 * c + x1 * s;
  float oo = -x0 * s + x1 * c;
  int c0 = 2 * i;
  int g = c0 >> 6, d = c0 & 63;
  _Float16* dst = kh + (((size_t)b * NG + g) * TT + t) * HH + d;
  dst[0] = (_Float16)oe;
  dst[1] = (_Float16)oo;
}

// v stored transposed: vt[(b,g)][d][t]
__global__ void pack_v_kernel(const _Float16* __restrict__ qkv,
                              _Float16* __restrict__ vt) {
  size_t idx = (size_t)blockIdx.x * blockDim.x + threadIdx.x;
  if (idx >= (size_t)BB * TT * GHH) return;
  int c = (int)(idx & 255);
  size_t bt = idx >> 8;
  int t = (int)(bt & (TT - 1)); int b = (int)(bt >> 10);
  int g = c >> 6, d = c & 63;
  vt[(((size_t)b * NG + g) * HH + d) * TT + t] = qkv[bt * NW + 768 + c];
}

// ---------------------------------------------------------------------------
// 4) flash attention: 4-wave blocks, each block owns 64 Q rows of one (b,g,p);
//    K/V j-tiles (32 wide) staged to LDS with double-buffered async DMA,
//    online softmax in lane-replicated registers (shfl over 16-lane groups).
// ---------------------------------------------------------------------------
__global__ __launch_bounds__(128) void attn_kernel(const _Float16* __restrict__ qh,
                                                   const _Float16* __restrict__ kh,
                                                   const _Float16* __restrict__ vt,
                                                   float* __restrict__ out) {
  __shared__ _Float16 Kbuf[2][32 * HH];   // rows j0..j0+31, 64-wide  (4 KB each)
  __shared__ _Float16 Vbuf[2][HH * 32];   // rows d=0..63, 32 j-cols  (4 KB each)
  __shared__ _Float16 Pbuf[4][16 * 32];   // per-wave P bounce        (1 KB each)

  int wid = blockIdx.x;
  int it = wid & 15; wid >>= 4;       // T/64 blocks
  int p  = wid & 1;  wid >>= 1;
  int g  = wid & 3;  wid >>= 2;
  int b  = wid;

  int tid  = threadIdx.x;             // 0..127
  int wv   = tid >> 5;                // wave in block
  int lane = tid & 31;
  int half = lane >> 4, col = lane & 15;
  int i0   = it * 64 + wv * 16;       // this wave's Q rows
  int jmax = it * 64 + 63;            // causal extent of the block

  const _Float16* qrow = qh + ((((size_t)b * NG + g) * NPG + p) * TT + i0) * HH;
  const _Float16* krow = kh + (((size_t)b * NG + g) * TT) * HH;
  const _Float16* vrow = vt + (((size_t)b * NG + g) * HH) * TT;

  v16h aq0 = load_a16(qrow, HH);
  v16h aq1 = load_a16(qrow + 32, HH);

  v8f acc0 = {}, acc1 = {}, acc2 = {}, acc3 = {};
  float m_s[8], l_s[8];
#pragma unroll
  for (int r = 0; r < 8; ++r) { m_s[r] = -1e30f; l_s[r] = 0.0f; }

  // ---- stage one j-tile (4 async ops per lane: 2 for K, 2 for V) ----
  auto stage = [&](int j0, int nbuf) {
    const char* kg = (const char*)(krow + (size_t)j0 * HH);   // 4 KB contiguous
    char* kl = (char*)Kbuf[nbuf];
#pragma unroll
    for (int q = 0; q < 2; ++q) {
      int off = (q * 128 + tid) * 16;
      ASYNC_CP16(kl + off, kg + off);
    }
#pragma unroll
    for (int q = 0; q < 2; ++q) {
      int idx = q * 128 + tid;                  // 0..255
      int d = idx >> 2, part = idx & 3;         // 64 rows x 4 x 16B
      const char* vg = (const char*)(vrow + (size_t)d * TT + j0) + part * 16;
      char* vl = (char*)Vbuf[nbuf] + d * 64 + part * 16;
      ASYNC_CP16(vl, vg);
    }
  };

  stage(0, 0);

  for (int j0 = 0; j0 <= jmax; j0 += 32) {
    int bi = (j0 >> 5) & 1;
    bool has_next = (j0 + 32 <= jmax);
    if (has_next) stage(j0 + 32, bi ^ 1);   // overlap next tile's DMA
    if (has_next) { WAIT_ASYNC(4); } else { WAIT_ASYNC(0); }
    __syncthreads();                        // staged tile visible block-wide

    // ---- S = Q K^T (two 16-col tiles from LDS K) ----
    const _Float16* kb = Kbuf[bi];
    v8f s0 = {}, s1 = {};
    {
      v16h bk;
      bk = load_b16(kb, HH);                s0 = wmma16(aq0, bk, s0);
      bk = load_b16(kb + 32, HH);           s0 = wmma16(aq1, bk, s0);
      bk = load_b16(kb + 16 * HH, HH);      s1 = wmma16(aq0, bk, s1);
      bk = load_b16(kb + 16 * HH + 32, HH); s1 = wmma16(aq1, bk, s1);
    }
#pragma unroll
    for (int e = 0; e < 8; ++e) { s0[e] *= 0.125f; s1[e] *= 0.125f; }

    if (j0 + 31 > i0) {   // causal mask (covers fully-masked tiles too)
#pragma unroll
      for (int e = 0; e < 8; ++e) {
        int i = i0 + e + 8 * half;
        if (j0 + col      > i) s0[e] = -1e30f;
        if (j0 + 16 + col > i) s1[e] = -1e30f;
      }
    }

    // ---- online softmax ----
    float alpha[8];
#pragma unroll
    for (int e = 0; e < 8; ++e) {
      float mx = fmaxf(s0[e], s1[e]);
      mx = fmaxf(mx, __shfl_xor(mx, 1));
      mx = fmaxf(mx, __shfl_xor(mx, 2));
      mx = fmaxf(mx, __shfl_xor(mx, 4));
      mx = fmaxf(mx, __shfl_xor(mx, 8));
      float mn = fmaxf(m_s[e], mx);
      alpha[e] = __expf(m_s[e] - mn);
      m_s[e] = mn;
      s0[e] = __expf(s0[e] - mn);
      s1[e] = __expf(s1[e] - mn);
      float rs = s0[e] + s1[e];
      rs += __shfl_xor(rs, 1);
      rs += __shfl_xor(rs, 2);
      rs += __shfl_xor(rs, 4);
      rs += __shfl_xor(rs, 8);
      l_s[e] = l_s[e] * alpha[e] + rs;
      acc0[e] *= alpha[e]; acc1[e] *= alpha[e];
      acc2[e] *= alpha[e]; acc3[e] *= alpha[e];
    }

    // ---- C-layout P -> A-layout via per-wave LDS bounce ----
    _Float16* pb = Pbuf[wv];
#pragma unroll
    for (int e = 0; e < 8; ++e) {
      int row = e + 8 * half;
      pb[row * 32 + col]      = (_Float16)s0[e];
      pb[row * 32 + 16 + col] = (_Float16)s1[e];
    }
    __syncthreads();
    v16h ap = load_a16(pb, 32);

    // ---- O += P V (h=64 as 4 n-tiles, V from LDS) ----
    const _Float16* vb = Vbuf[bi];
    acc0 = wmma16(ap, load_b16(vb + 0 * 16 * 32, 32), acc0);
    acc1 = wmma16(ap, load_b16(vb + 1 * 16 * 32, 32), acc1);
    acc2 = wmma16(ap, load_b16(vb + 2 * 16 * 32, 32), acc2);
    acc3 = wmma16(ap, load_b16(vb + 3 * 16 * 32, 32), acc3);
    __syncthreads();   // all reads of buf[bi] done before it is re-staged
  }

  // epilogue: normalize, write in the reference's (b,p,g,i,h) flattening
  float* obase = out + (size_t)b * TT * DD + (((size_t)p * NG + g) * TT + i0) * HH;
#pragma unroll
  for (int e = 0; e < 8; ++e) {
    float inv = 1.0f / l_s[e];
    float* orow = obase + (size_t)(e + 8 * half) * HH;
    orow[col]      = acc0[e] * inv;
    orow[16 + col] = acc1[e] * inv;
    orow[32 + col] = acc2[e] * inv;
    orow[48 + col] = acc3[e] * inv;
  }
}

// ---------------------------------------------------------------------------
// host launcher
// ---------------------------------------------------------------------------
extern "C" void kernel_launch(void* const* d_in, const int* in_sizes, int n_in,
                              void* d_out, int out_size, void* d_ws, size_t ws_size,
                              hipStream_t stream) {
  const float* x  = (const float*)d_in[0];
  const float* Wq = (const float*)d_in[1];
  const float* Wk = (const float*)d_in[2];
  const float* Wv = (const float*)d_in[3];
  float* out = (float*)d_out;

  char* ws = (char*)d_ws;
  size_t off = 0;
  _Float16* xh  = (_Float16*)(ws + off); off += (size_t)BB * TT * DD * 2;
  _Float16* wt  = (_Float16*)(ws + off); off += (size_t)NW * DD * 2;
  _Float16* qkv = (_Float16*)(ws + off); off += (size_t)BB * TT * NW * 2;
  _Float16* qh  = (_Float16*)(ws + off); off += (size_t)BB * TT * DD * 2;
  _Float16* kh  = (_Float16*)(ws + off); off += (size_t)BB * TT * GHH * 2;
  _Float16* vt  = (_Float16*)(ws + off); off += (size_t)BB * TT * GHH * 2;
  float* cq = (float*)(ws + off); off += (size_t)TT * 256 * 4;
  float* sq = (float*)(ws + off); off += (size_t)TT * 256 * 4;
  float* ck = (float*)(ws + off); off += (size_t)TT * 128 * 4;
  float* sk = (float*)(ws + off); off += (size_t)TT * 128 * 4;
  (void)ws_size; (void)in_sizes; (void)n_in; (void)out_size;

  cast_x_kernel<<<(BB * TT * DD) / 256, 256, 0, stream>>>(x, xh);
  cast_w_kernel<<<(NW * DD) / 256, 256, 0, stream>>>(Wq, Wk, Wv, wt);
  tables_kernel<<<(TT * 384) / 256, 256, 0, stream>>>(cq, sq, ck, sk);

  // 1024 m-tiles * 16 n-blocks = 16384 waves, 4 waves/block
  proj_kernel<<<4096, 128, 0, stream>>>(xh, wt, qkv);

  rope_q_kernel<<<(BB * TT * 256) / 256, 256, 0, stream>>>(qkv, cq, sq, qh);
  rope_k_kernel<<<(BB * TT * 128) / 256, 256, 0, stream>>>(qkv, ck, sk, kh);
  pack_v_kernel<<<(BB * TT * 256) / 256, 256, 0, stream>>>(qkv, vt);

  // B*NG*NPG*(T/64) = 2048 blocks of 4 waves
  attn_kernel<<<BB * NG * NPG * (TT / 64), 128, 0, stream>>>(qh, kh, vt, out);
}
